// PostNormMamba_68564857913689
// MI455X (gfx1250) — matrix operand order repaired and checked by reading, not compiled, not run on hardware
//
#include <hip/hip_runtime.h>
#include <hip/hip_bf16.h>
#include <math.h>

// ---------------------------------------------------------------------------
// PostNorm Mamba block for MI455X (gfx1250, wave32, WMMA + TDM).
// GEMMs: bf16 A/B, f32 accumulate via v_wmma_f32_16x16x32_bf16, tiles staged
//        into LDS by the Tensor Data Mover (tensor_load_to_lds), double-buffered.
// Scan:  fp32, 1 thread per (b, d_inner) channel, 16 states in registers.
// ---------------------------------------------------------------------------

#define B_SZ    4
#define LSEQ    2048
#define DMODEL  512
#define DINNER  1024
#define DSTATE  16
#define DCONV   4
#define MROWS   (B_SZ * LSEQ)        // 8192

typedef __attribute__((ext_vector_type(16))) __bf16       bf16x16;
typedef __attribute__((ext_vector_type(8)))  float        f32x8;
typedef __attribute__((ext_vector_type(4)))  unsigned int u32x4;
typedef __attribute__((ext_vector_type(8)))  int          i32x8;
typedef __attribute__((ext_vector_type(4)))  int          i32x4;

#if __has_builtin(__builtin_amdgcn_tensor_load_to_lds) && \
    __has_builtin(__builtin_amdgcn_s_wait_tensorcnt)
#define HAVE_TDM 1
#define NBUF 2
#else
#define HAVE_TDM 0
#define NBUF 1
#endif

// ---------------------------------------------------------------------------
// Generic fp32 -> bf16 conversion
// ---------------------------------------------------------------------------
__global__ void __launch_bounds__(256) cvt_f32_bf16(const float* __restrict__ src,
                                                    __bf16* __restrict__ dst, int n) {
    int i = blockIdx.x * 256 + threadIdx.x;
    if (i < n) dst[i] = (__bf16)src[i];
}

#if HAVE_TDM
// ---------------------------------------------------------------------------
// Issue one TDM 2D tile load: rows x 32 bf16 tile from global (row stride =
// ld elements) into LDS at lds_off, padded +8 bf16 (16B) per 32-element (64B)
// row so the LDS image matches a [rows][40] bf16 array.
// D# packing per CDNA5 ISA ch.8: group0 = {count/lds_addr/global_addr/type},
// group1 = {data_size,pad, tensor dims, tile dims, dim0 stride}.
// This toolchain's builtin is the 6-arg form:
//   (u32x4 g0, i32x8 g1, i32x4 g2, i32x4 g3, i32x8 gx, i32 cpol)
// g2/g3/gx zero-filled: tile is 2D (tile_dim2=0), groups 2/3 unused.
// ---------------------------------------------------------------------------
__device__ __forceinline__ void tdm_load_tile_bf16(const __bf16* gsrc,
                                                   unsigned lds_off,
                                                   int rows, int ld) {
    const unsigned long long ga = (unsigned long long)(uintptr_t)gsrc;
    u32x4 g0;
    g0[0] = 1u;                                                  // count=1, user mode
    g0[1] = lds_off;                                             // lds_addr [63:32]
    g0[2] = (unsigned)(ga & 0xFFFFFFFFu);                        // global_addr lo
    g0[3] = (unsigned)((ga >> 32) & 0x01FFFFFFu) | (2u << 30);   // addr hi | type=2
    i32x8 g1;
    // data_size=1 (2B) | pad_enable | pad_interval=3 (16 DW = 64B row)
    // | pad_amount=3 (4 DW = 16B = 8 bf16)
    g1[0] = (int)((1u << 16) | (1u << 20) | (3u << 22) | (3u << 25));
    g1[1] = (int)(((unsigned)ld & 0xFFFFu) << 16);               // tensor_dim0 = ld
    g1[2] = (int)((((unsigned)ld >> 16) & 0xFFFFu) | (0xFFFFu << 16)); // tdim1 lo
    g1[3] = (int)(0xFu | (32u << 16));                           // tdim1 hi, tile_dim0=32
    g1[4] = rows;                                                // tile_dim1, tile_dim2=0
    g1[5] = ld;                                                  // dim0_stride lo32
    g1[6] = 0;                                                   // dim0_stride hi16
    g1[7] = 0;
    const i32x4 z4 = {0, 0, 0, 0};
    const i32x8 z8 = {0, 0, 0, 0, 0, 0, 0, 0};
    __builtin_amdgcn_tensor_load_to_lds(g0, g1, z4, z4, z8, 0);
}
#endif

// ---------------------------------------------------------------------------
// Tiled WMMA GEMM:  C[M,N] = A[M,K] * B[N,K]^T  (B row-major over N = weights)
// Block tile 128x64, BK=32, 256 threads = 8 waves, each wave a 32x32 tile
// (2x2 accumulators of v_wmma_f32_16x16x32_bf16).
// TDM path: wave 0 DMAs the next A/B tiles into the spare LDS buffer while all
// waves run WMMA on the current one; completion via s_wait_tensorcnt + barrier.
// Requires: M%128==0, N%64==0, K%32==0, lda/ldb multiples of 8.
// ---------------------------------------------------------------------------
__global__ void __launch_bounds__(256) gemm_bf16_wmma(
    const __bf16* __restrict__ A, int lda,
    const __bf16* __restrict__ Bw, int ldb,
    float* __restrict__ C, int ldc, int K)
{
    __shared__ __bf16 sA[NBUF][128][40];   // 32 + 8 pad (bank-conflict mitigation)
    __shared__ __bf16 sB[NBUF][64][40];

    const int tid  = threadIdx.x;
    const int lane = tid & 31;
    const int wave = tid >> 5;
    const int wm   = (wave & 3) * 32;     // wave M offset within block tile
    const int wn   = (wave >> 2) * 32;    // wave N offset within block tile
    const int m0   = blockIdx.x * 128;
    const int n0   = blockIdx.y * 64;

    f32x8 acc[2][2] = {};

    const int r    = lane & 15;     // row/col within 16x16 tile
    const int hsel = lane >> 4;     // lane-half select

#if HAVE_TDM
    const bool issuer = (wave == 0);
    if (issuer) {
        tdm_load_tile_bf16(A  + (size_t)m0 * lda,
                           (unsigned)(uintptr_t)&sA[0][0][0], 128, lda);
        tdm_load_tile_bf16(Bw + (size_t)n0 * ldb,
                           (unsigned)(uintptr_t)&sB[0][0][0], 64, ldb);
    }
    int p = 0;
#else
    const int arow = tid >> 1, acol = (tid & 1) * 16;   // A: 128 rows x 32 cols
    const int brow = tid >> 2, bcol = (tid & 3) * 8;    // B: 64 rows x 32 cols
#endif

    for (int k0 = 0; k0 < K; k0 += 32) {
#if HAVE_TDM
        if (issuer) {
            if (k0 + 32 < K) {
                // prefetch next K tile into the spare buffer via TDM
                tdm_load_tile_bf16(A  + (size_t)m0 * lda + (k0 + 32),
                                   (unsigned)(uintptr_t)&sA[p ^ 1][0][0], 128, lda);
                tdm_load_tile_bf16(Bw + (size_t)n0 * ldb + (k0 + 32),
                                   (unsigned)(uintptr_t)&sB[p ^ 1][0][0], 64, ldb);
                __builtin_amdgcn_s_wait_tensorcnt(2);   // current pair complete
            } else {
                __builtin_amdgcn_s_wait_tensorcnt(0);   // drain
            }
        }
        __syncthreads();                                 // publish LDS tile to all waves
        const __bf16 (*tA)[40] = sA[p];
        const __bf16 (*tB)[40] = sB[p];
#else
        const __bf16* ga = A  + (size_t)(m0 + arow) * lda + k0 + acol;
        const __bf16* gb = Bw + (size_t)(n0 + brow) * ldb + k0 + bcol;
        *(uint4*)&sA[0][arow][acol]     = *(const uint4*)ga;
        *(uint4*)&sA[0][arow][acol + 8] = *(const uint4*)(ga + 8);
        *(uint4*)&sB[0][brow][bcol]     = *(const uint4*)gb;
        if (k0 + 32 < K) {
            __builtin_prefetch(ga + 32, 0, 1);
            __builtin_prefetch(gb + 32, 0, 1);
        }
        __syncthreads();
        const __bf16 (*tA)[40] = sA[0];
        const __bf16 (*tB)[40] = sB[0];
#endif

        // A fragments: lane(0-15) holds K {0..7,16..23}, lane(16-31) holds K {8..15,24..31}
        bf16x16 af[2];
        #pragma unroll
        for (int mi = 0; mi < 2; ++mi) {
            const int row = wm + mi * 16 + r;
            const int kh  = hsel * 8;
            ((uint4*)&af[mi])[0] = *(const uint4*)&tA[row][kh];
            ((uint4*)&af[mi])[1] = *(const uint4*)&tA[row][16 + kh];
        }
        // B fragments: lane(0-15) holds K {0..7,16..23}, lane(16-31) holds K {8..15,24..31}
        bf16x16 bfr[2];
        #pragma unroll
        for (int ni = 0; ni < 2; ++ni) {
            const int col = wn + ni * 16 + r;
            const int kb  = hsel * 8;
            ((uint4*)&bfr[ni])[0] = *(const uint4*)&tB[col][kb];
            ((uint4*)&bfr[ni])[1] = *(const uint4*)&tB[col][kb + 16];
        }
        #pragma unroll
        for (int mi = 0; mi < 2; ++mi)
            #pragma unroll
            for (int ni = 0; ni < 2; ++ni)
                acc[mi][ni] = __builtin_amdgcn_wmma_f32_16x16x32_bf16(
                    false, af[mi], false, bfr[ni], (short)0, acc[mi][ni], false, false);
        __syncthreads();
#if HAVE_TDM
        p ^= 1;
#endif
    }

    // C/D layout: VGPR e, lanes 0-15 -> (M=e, N=lane); lanes 16-31 -> (M=8+e, N=lane-16)
    #pragma unroll
    for (int mi = 0; mi < 2; ++mi)
        #pragma unroll
        for (int ni = 0; ni < 2; ++ni) {
            const size_t cn = (size_t)(n0 + wn + ni * 16 + r);
            const int    cm = m0 + wm + mi * 16 + hsel * 8;
            #pragma unroll
            for (int e = 0; e < 8; ++e)
                C[(size_t)(cm + e) * ldc + cn] = acc[mi][ni][e];
        }
}

// ---------------------------------------------------------------------------
// Causal depthwise conv(4) + SiLU over u = xz[:, :DINNER]; emits f32 and bf16
// ---------------------------------------------------------------------------
__global__ void __launch_bounds__(256) conv_silu(
    const float* __restrict__ xz, const float* __restrict__ cw,
    const float* __restrict__ cb, float* __restrict__ uf, __bf16* __restrict__ ub)
{
    const int idx = blockIdx.x * 256 + threadIdx.x;   // MROWS*DINNER threads
    const int m = idx >> 10, d = idx & (DINNER - 1);
    const int l = m & (LSEQ - 1);
    float acc = cb[d];
    #pragma unroll
    for (int w = 0; w < DCONV; ++w) {
        const int ll = l - (DCONV - 1) + w;
        if (ll >= 0)
            acc += cw[d * DCONV + w] * xz[(size_t)(m - (DCONV - 1) + w) * (2 * DINNER) + d];
    }
    const float s = acc / (1.f + __expf(-acc));
    uf[idx] = s;
    ub[idx] = (__bf16)s;
}

// ---------------------------------------------------------------------------
// delta = softplus(gemm_out + dt_proj_b), in place
// ---------------------------------------------------------------------------
__global__ void __launch_bounds__(256) softplus_bias(float* __restrict__ g,
                                                     const float* __restrict__ bias)
{
    const int idx = blockIdx.x * 256 + threadIdx.x;
    const int d = idx & (DINNER - 1);
    const float v = g[idx] + bias[d];
    g[idx] = (v > 20.f) ? v : log1pf(__expf(v));
}

// ---------------------------------------------------------------------------
// Selective scan: 1 thread per (b, d). 16 states in registers, B_t/C_t via LDS.
// y includes the D*u skip term.
// ---------------------------------------------------------------------------
__global__ void __launch_bounds__(128) selective_scan(
    const float* __restrict__ delta, const float* __restrict__ uf,
    const float* __restrict__ dbc, const float* __restrict__ A_log,
    const float* __restrict__ Dv, float* __restrict__ y)
{
    __shared__ float bc[2 * DSTATE];
    const int b = blockIdx.x >> 3;                    // DINNER/128 = 8 blocks per batch
    const int d = ((blockIdx.x & 7) << 7) + threadIdx.x;

    float Areg[DSTATE], h[DSTATE];
    #pragma unroll
    for (int s = 0; s < DSTATE; ++s) {
        Areg[s] = -__expf(A_log[d * DSTATE + s]);
        h[s] = 0.f;
    }
    const float Dd = Dv[d];

    for (int t = 0; t < LSEQ; ++t) {
        const int row = b * LSEQ + t;
        __syncthreads();
        if (threadIdx.x < 2 * DSTATE)
            bc[threadIdx.x] = dbc[(size_t)row * 64 + 32 + threadIdx.x];
        __syncthreads();
        const float dt = delta[(size_t)row * DINNER + d];
        const float ut = uf[(size_t)row * DINNER + d];
        const float du = dt * ut;
        float acc = 0.f;
        #pragma unroll
        for (int s = 0; s < DSTATE; ++s) {
            const float dA = __expf(dt * Areg[s]);
            h[s] = dA * h[s] + du * bc[s];
            acc += h[s] * bc[DSTATE + s];
        }
        y[(size_t)row * DINNER + d] = acc + Dd * ut;
    }
}

// ---------------------------------------------------------------------------
// Gate: yg = y * silu(z), emitted directly as bf16 (A operand of out_proj)
// ---------------------------------------------------------------------------
__global__ void __launch_bounds__(256) gate_silu(
    const float* __restrict__ y, const float* __restrict__ xz, __bf16* __restrict__ yg)
{
    const int idx = blockIdx.x * 256 + threadIdx.x;
    const int m = idx >> 10, d = idx & (DINNER - 1);
    const float z = xz[(size_t)m * (2 * DINNER) + DINNER + d];
    const float g = z / (1.f + __expf(-z));
    yg[idx] = (__bf16)(y[idx] * g);
}

// ---------------------------------------------------------------------------
// RMSNorm over DMODEL + residual. One block (256 threads) per row.
// ---------------------------------------------------------------------------
__global__ void __launch_bounds__(256) rmsnorm_res(
    const float* __restrict__ op, const float* __restrict__ x,
    const float* __restrict__ w, float* __restrict__ out)
{
    __shared__ float red[256];
    __shared__ float sscale;
    const int m = blockIdx.x, t = threadIdx.x;
    const size_t base = (size_t)m * DMODEL;
    const float v0 = op[base + t];
    const float v1 = op[base + 256 + t];
    red[t] = v0 * v0 + v1 * v1;
    __syncthreads();
    #pragma unroll
    for (int s = 128; s > 0; s >>= 1) {
        if (t < s) red[t] += red[t + s];
        __syncthreads();
    }
    if (t == 0) sscale = rsqrtf(red[0] / (float)DMODEL + 1e-5f);
    __syncthreads();
    const float sc = sscale;
    out[base + t]       = v0 * sc * w[t]       + x[base + t];
    out[base + 256 + t] = v1 * sc * w[256 + t] + x[base + 256 + t];
}

// ---------------------------------------------------------------------------
// Host launcher
// ---------------------------------------------------------------------------
extern "C" void kernel_launch(void* const* d_in, const int* in_sizes, int n_in,
                              void* d_out, int out_size, void* d_ws, size_t ws_size,
                              hipStream_t stream) {
    (void)in_sizes; (void)n_in; (void)out_size; (void)ws_size;

    const float* x         = (const float*)d_in[0];   // [B,L,512]
    const float* in_proj_w = (const float*)d_in[1];   // [2048,512]
    const float* conv_w    = (const float*)d_in[2];   // [1024,1,4]
    const float* conv_b    = (const float*)d_in[3];   // [1024]
    const float* x_proj_w  = (const float*)d_in[4];   // [64,1024]
    const float* dt_proj_w = (const float*)d_in[5];   // [1024,32]
    const float* dt_proj_b = (const float*)d_in[6];   // [1024]
    const float* A_log     = (const float*)d_in[7];   // [1024,16]
    const float* Dv        = (const float*)d_in[8];   // [1024]
    const float* out_w     = (const float*)d_in[9];   // [512,1024]
    const float* rms_w     = (const float*)d_in[10];  // [512]
    float* out = (float*)d_out;

    // workspace carve-out (256B aligned)
    char* ws = (char*)d_ws;
    size_t off = 0;
    auto carve = [&](size_t bytes) {
        size_t o = off;
        off += (bytes + 255) & ~(size_t)255;
        return (void*)(ws + o);
    };
    __bf16* xbf     = (__bf16*)carve((size_t)MROWS * DMODEL * 2);        // 8 MB
    __bf16* w_in_bf = (__bf16*)carve((size_t)2 * DINNER * DMODEL * 2);   // 2 MB
    __bf16* w_xp_bf = (__bf16*)carve((size_t)64 * DINNER * 2);           // 128 KB
    __bf16* w_dt_bf = (__bf16*)carve((size_t)DINNER * 32 * 2);           // 64 KB
    __bf16* w_ot_bf = (__bf16*)carve((size_t)DMODEL * DINNER * 2);       // 1 MB
    float*  xz      = (float*) carve((size_t)MROWS * 2 * DINNER * 4);    // 64 MB
    float*  uf      = (float*) carve((size_t)MROWS * DINNER * 4);        // 32 MB
    __bf16* ubf     = (__bf16*)carve((size_t)MROWS * DINNER * 2);        // 16 MB
    float*  dbc     = (float*) carve((size_t)MROWS * 64 * 4);            // 2 MB
    __bf16* dbc_bf  = (__bf16*)carve((size_t)MROWS * 64 * 2);            // 1 MB
    float*  delta   = (float*) carve((size_t)MROWS * DINNER * 4);        // 32 MB
    float*  ybuf    = (float*) carve((size_t)MROWS * DINNER * 4);        // 32 MB
    __bf16* ygbf    = (__bf16*)carve((size_t)MROWS * DINNER * 2);        // 16 MB
    float*  outp    = (float*) carve((size_t)MROWS * DMODEL * 4);        // 16 MB

    auto cvt = [&](const float* s, __bf16* d, int n) {
        cvt_f32_bf16<<<(n + 255) / 256, 256, 0, stream>>>(s, d, n);
    };

    // 1) bf16 conversions of GEMM operands
    cvt(x,         xbf,     MROWS * DMODEL);
    cvt(in_proj_w, w_in_bf, 2 * DINNER * DMODEL);
    cvt(x_proj_w,  w_xp_bf, 64 * DINNER);
    cvt(dt_proj_w, w_dt_bf, DINNER * 32);
    cvt(out_w,     w_ot_bf, DMODEL * DINNER);

    // 2) in_proj: xz[8192,2048] = x[8192,512] @ in_proj_w^T
    gemm_bf16_wmma<<<dim3(MROWS / 128, (2 * DINNER) / 64), 256, 0, stream>>>(
        xbf, DMODEL, w_in_bf, DMODEL, xz, 2 * DINNER, DMODEL);

    // 3) depthwise causal conv + SiLU -> u (f32 + bf16)
    conv_silu<<<(MROWS * DINNER) / 256, 256, 0, stream>>>(xz, conv_w, conv_b, uf, ubf);

    // 4) x_proj: dbc[8192,64] = u @ x_proj_w^T
    gemm_bf16_wmma<<<dim3(MROWS / 128, 1), 256, 0, stream>>>(
        ubf, DINNER, w_xp_bf, DINNER, dbc, 64, DINNER);
    cvt(dbc, dbc_bf, MROWS * 64);

    // 5) dt_proj: delta_raw[8192,1024] = dbc[:, :32] @ dt_proj_w^T  (lda=64, K=32)
    gemm_bf16_wmma<<<dim3(MROWS / 128, DINNER / 64), 256, 0, stream>>>(
        dbc_bf, 64, w_dt_bf, 32, delta, DINNER, 32);
    softplus_bias<<<(MROWS * DINNER) / 256, 256, 0, stream>>>(delta, dt_proj_b);

    // 6) selective scan (includes D*u skip)
    selective_scan<<<B_SZ * (DINNER / 128), 128, 0, stream>>>(
        delta, uf, dbc, A_log, Dv, ybuf);

    // 7) gate with silu(z), emit bf16 A for out_proj
    gate_silu<<<(MROWS * DINNER) / 256, 256, 0, stream>>>(ybuf, xz, ygbf);

    // 8) out_proj: outp[8192,512] = yg @ out_proj_w^T
    gemm_bf16_wmma<<<dim3(MROWS / 128, DMODEL / 64), 256, 0, stream>>>(
        ygbf, DINNER, w_ot_bf, DINNER, outp, DMODEL, DINNER);

    // 9) RMSNorm + residual
    rmsnorm_res<<<MROWS, 256, 0, stream>>>(outp, x, rms_w, out);
}
